// SwitchRouter_35871566856544
// MI455X (gfx1250) — matrix operand order repaired and checked by reading, not compiled, not run on hardware
//
#include <hip/hip_runtime.h>
#include <math.h>

// ---------------- problem constants (B=8,S=2048,D=2048,E=64) ----------------
#define NTOK     16384              // B*S
#define DDIM     2048
#define NEXP     64
#define CAPACITY 281                // int(16384*1.1/64)
#define Z_COEF   0.001f

#define TILE_M   64                 // tokens per block
#define KC       64                 // K chunk staged in LDS
#define XSTR     72                 // bf16 row stride (16B-aligned, padded)
#define LSTR     68                 // f32 logits row stride (padded)
#define SLICES   16                 // parallel scan slices for rank kernel

typedef __bf16 bf16_t;
typedef __attribute__((ext_vector_type(16))) __bf16 v16bf;
typedef __attribute__((ext_vector_type(8)))  __bf16 v8bf;
typedef __attribute__((ext_vector_type(8)))  float  v8f;

union Frag16 { v16bf v; v8bf h[2]; };

// ---------------------------------------------------------------------------
// Kernel 1: logits GEMM (bf16 WMMA, f32 accum) fused with softmax / argmax /
// z-loss partial / per-expert prob-sum partial.
// Block = 128 threads (4 waves); each wave computes a 16x64 logits tile.
// ---------------------------------------------------------------------------
__global__ __launch_bounds__(128)
void router_gemm_softmax(const float* __restrict__ x, const float* __restrict__ W,
                         float* __restrict__ probs_out, float* __restrict__ flat_p,
                         int* __restrict__ flat_e, float* __restrict__ prob_sum,
                         float* __restrict__ z_sum)
{
    __shared__ __align__(16) bf16_t xs [TILE_M * XSTR];   // token tile (bf16)
    __shared__ __align__(16) bf16_t wsh[NEXP  * XSTR];    // all 64 experts (bf16)
    __shared__ float lg[TILE_M * LSTR];                   // logits tile (f32)
    __shared__ float lse_s[TILE_M];
    __shared__ float red[TILE_M];

    const int tid  = threadIdx.x;
    const int tok0 = blockIdx.x * TILE_M;
    const int wv   = tid >> 5;          // wave id 0..3 -> token sub-tile
    const int lane = tid & 31;
    const int m    = lane & 15;         // row (A) / col (B,C)
    const int hh   = lane >> 4;         // half-wave select

    v8f acc[4] = {};                    // 4 expert tiles of 16x16 f32

    for (int k0 = 0; k0 < DDIM; k0 += KC) {
        __syncthreads();
        // stage x tile + W tile to LDS, f32 -> bf16 on the fly (coalesced)
        for (int idx = tid; idx < TILE_M * KC; idx += 128) {
            int r = idx >> 6, c = idx & (KC - 1);
            xs [r * XSTR + c] = (bf16_t)x[(size_t)(tok0 + r) * DDIM + k0 + c];
            wsh[r * XSTR + c] = (bf16_t)W[(size_t)r          * DDIM + k0 + c];
        }
        if (k0 + KC < DDIM)  // hint next x chunk into cache (global_prefetch_b8)
            __builtin_prefetch(&x[(size_t)(tok0 + (tid & 63)) * DDIM + k0 + KC], 0, 1);
        __syncthreads();

        #pragma unroll
        for (int ks = 0; ks < KC; ks += 32) {
            // A frag: 16x32 bf16, lane m+16h holds K = [16h, 16h+16) contiguous
            Frag16 a;
            const bf16_t* xr = &xs[(16 * wv + m) * XSTR + ks];
            a.h[0] = *(const v8bf*)(xr + 16 * hh);
            a.h[1] = *(const v8bf*)(xr + 16 * hh + 8);
            #pragma unroll
            for (int t = 0; t < 4; ++t) {
                // B frag: 32x16 bf16, lane n+16h holds K = [16h, 16h+16) contiguous
                Frag16 b;
                const bf16_t* wr = &wsh[(16 * t + m) * XSTR + ks + 16 * hh];
                b.h[0] = *(const v8bf*)(wr);
                b.h[1] = *(const v8bf*)(wr + 8);
                acc[t] = __builtin_amdgcn_wmma_f32_16x16x32_bf16(
                    false, a.v, false, b.v, (short)0, acc[t], false, false);
            }
        }
    }

    // spill logits tile to LDS (C layout: vgpr r -> row r+8h, col = lane%16)
    __syncthreads();
    #pragma unroll
    for (int t = 0; t < 4; ++t)
        #pragma unroll
        for (int r = 0; r < 8; ++r)
            lg[(16 * wv + r + 8 * hh) * LSTR + 16 * t + m] = acc[t][r];
    __syncthreads();

    // softmax / argmax / z-loss per token (one thread per token)
    if (tid < TILE_M) {
        const float* row = &lg[tid * LSTR];
        float mx = row[0];
        for (int e = 1; e < NEXP; ++e) mx = fmaxf(mx, row[e]);
        float s = 0.f;
        for (int e = 0; e < NEXP; ++e) s += __expf(row[e] - mx);
        const float lse = mx + __logf(s);
        lse_s[tid] = lse;
        const int g = tok0 + tid;
        float best = -1.f; int be = 0;
        for (int e = 0; e < NEXP; ++e) {
            float pr = __expf(row[e] - lse);
            probs_out[(size_t)g * NEXP + e] = pr;
            if (pr > best) { best = pr; be = e; }   // first max (matches argmax)
        }
        flat_p[g] = best;
        flat_e[g] = be;
        red[tid]  = lse * lse;                       // z-loss contribution
    }
    __syncthreads();
    if (tid == 0) {
        float zs = 0.f;
        for (int i = 0; i < TILE_M; ++i) zs += red[i];
        atomicAdd(z_sum, zs);
    }
    // per-expert prob sums (one thread per expert over the 64 tokens here)
    if (tid < NEXP) {
        float s = 0.f;
        for (int tr = 0; tr < TILE_M; ++tr)
            s += __expf(lg[tr * LSTR + tid] - lse_s[tr]);
        atomicAdd(&prob_sum[tid], s);
    }
}

// ---------------------------------------------------------------------------
// Kernel 2: partial priority-rank. rank[i] = #{j in slice: e_j==e_i &&
// (p_j>p_i || (p_j==p_i && j<i))}  (== position under stable argsort(-p)).
// grid = (NTOK/256, SLICES); slices accumulate via atomics.
// ---------------------------------------------------------------------------
__global__ __launch_bounds__(256)
void rank_partial(const float* __restrict__ flat_p, const int* __restrict__ flat_e,
                  int* __restrict__ rank)
{
    __shared__ float sp[256];
    __shared__ int   se[256];
    const int tid = threadIdx.x;
    const int i   = blockIdx.x * 256 + tid;
    const float pi = flat_p[i];
    const int   ei = flat_e[i];
    const int start = blockIdx.y * (NTOK / SLICES);
    const int stop  = start + (NTOK / SLICES);
    int cnt = 0;
    for (int base = start; base < stop; base += 256) {
        __syncthreads();
        sp[tid] = flat_p[base + tid];
        se[tid] = flat_e[base + tid];
        __syncthreads();
        #pragma unroll 8
        for (int j = 0; j < 256; ++j) {
            const int gj = base + j;
            if (se[j] == ei && (sp[j] > pi || (sp[j] == pi && gj < i))) ++cnt;
        }
    }
    if (cnt) atomicAdd(&rank[i], cnt);
}

// ---------------------------------------------------------------------------
// Kernel 3: keep mask + expert usage histogram.
// ---------------------------------------------------------------------------
__global__ __launch_bounds__(256)
void keep_kernel(const int* __restrict__ rank, const int* __restrict__ flat_e,
                 int* __restrict__ keep, float* __restrict__ usage)
{
    __shared__ int hist[NEXP];
    const int tid = threadIdx.x;
    const int i   = blockIdx.x * 256 + tid;
    const int kp  = (rank[i] < CAPACITY) ? 1 : 0;
    keep[i] = kp;
    if (tid < NEXP) hist[tid] = 0;
    __syncthreads();
    atomicAdd(&hist[flat_e[i]], kp);
    __syncthreads();
    if (tid < NEXP && hist[tid] != 0) atomicAdd(&usage[tid], (float)hist[tid]);
}

// ---------------------------------------------------------------------------
// Kernel 4: dispatch + combine (identical) scatter.
// ---------------------------------------------------------------------------
__global__ __launch_bounds__(256)
void dispatch_kernel(const int* __restrict__ keep, const int* __restrict__ flat_e,
                     float* __restrict__ out)
{
    const int idx = blockIdx.x * 256 + threadIdx.x;   // [0, NTOK*NEXP)
    const int tok = idx >> 6, e = idx & 63;
    const float v = (keep[tok] && flat_e[tok] == e) ? 1.f : 0.f;
    out[idx] = v;                                      // dispatch
    out[(size_t)NTOK * NEXP + idx] = v;                // combine
}

// ---------------------------------------------------------------------------
// Kernel 5: aux loss = E * sum_e (prob_sum/N)*(usage/N) + Z_COEF * z_sum/N
// ---------------------------------------------------------------------------
__global__ void aux_kernel(const float* __restrict__ prob_sum,
                           const float* __restrict__ usage,
                           const float* __restrict__ z_sum,
                           float* __restrict__ out_aux)
{
    __shared__ float red[NEXP];
    const int e = threadIdx.x;
    red[e] = (prob_sum[e] / (float)NTOK) * (usage[e] / (float)NTOK);
    __syncthreads();
    if (e == 0) {
        float s = 0.f;
        for (int i = 0; i < NEXP; ++i) s += red[i];
        out_aux[0] = s * (float)NEXP + Z_COEF * (z_sum[0] / (float)NTOK);
    }
}

// ---------------------------------------------------------------------------
extern "C" void kernel_launch(void* const* d_in, const int* in_sizes, int n_in,
                              void* d_out, int out_size, void* d_ws, size_t ws_size,
                              hipStream_t stream)
{
    const float* x = (const float*)d_in[0];   // [N, D]
    const float* W = (const float*)d_in[1];   // [E, D]
    float* out = (float*)d_out;
    const size_t NE = (size_t)NTOK * NEXP;
    float* dispatch = out;                    // [N,E]
    float* probs    = out + 2 * NE;           // [N,E]
    float* aux      = out + 3 * NE;           // [1]

    // workspace layout
    char* ws = (char*)d_ws;
    float* flat_p   = (float*)ws;                       // N
    int*   flat_e   = (int*)  (ws + 1 * NTOK * 4);      // N
    int*   keep     = (int*)  (ws + 2 * NTOK * 4);      // N
    int*   rank     = (int*)  (ws + 3 * NTOK * 4);      // N  (needs zero)
    float* prob_sum = (float*)(ws + 4 * NTOK * 4);      // 64 (needs zero)
    float* usage    = prob_sum + NEXP;                  // 64 (needs zero)
    float* z_sum    = usage + NEXP;                     // 1  (needs zero)

    // zero the accumulated region: rank[N] + prob_sum/usage/z  (graph-capturable)
    hipMemsetAsync(rank, 0, (size_t)(NTOK + 2 * NEXP + 1) * sizeof(float), stream);

    router_gemm_softmax<<<NTOK / TILE_M, 128, 0, stream>>>(
        x, W, probs, flat_p, flat_e, prob_sum, z_sum);

    dim3 rg(NTOK / 256, SLICES);
    rank_partial<<<rg, 256, 0, stream>>>(flat_p, flat_e, rank);

    keep_kernel<<<NTOK / 256, 256, 0, stream>>>(rank, flat_e, keep, usage);

    dispatch_kernel<<<(NTOK * NEXP) / 256, 256, 0, stream>>>(keep, flat_e, dispatch);

    aux_kernel<<<1, NEXP, 0, stream>>>(prob_sum, usage, z_sum, aux);
}
